// SpeechDW_Former_50242527428697
// MI455X (gfx1250) — matrix operand order repaired and checked by reading, not compiled, not run on hardware
//
#include <hip/hip_runtime.h>
#include <hip/hip_bf16.h>

// Problem constants (from reference setup_inputs)
#define BB 4
#define TT 1024
#define CC 512
#define FF 2048
#define HH 8
#define DH 64

typedef __bf16 v16bf __attribute__((ext_vector_type(16)));
typedef __bf16 v8bf  __attribute__((ext_vector_type(8)));
typedef float  v8f   __attribute__((ext_vector_type(8)));

__device__ __forceinline__ unsigned short f2bf(float f) {
  unsigned u = __float_as_uint(f);
  unsigned r = u + 0x7FFFu + ((u >> 16) & 1u);   // round-to-nearest-even
  return (unsigned short)(r >> 16);
}

// pack 4 floats -> 4 bf16, one 8-byte LDS store (dst must be 8B aligned)
__device__ __forceinline__ void store4bf(unsigned short* dst, float4 f) {
  union { unsigned short us[4]; uint2 u2; } p;
  p.us[0] = f2bf(f.x); p.us[1] = f2bf(f.y);
  p.us[2] = f2bf(f.z); p.us[3] = f2bf(f.w);
  *(uint2*)dst = p.u2;
}

// two contiguous 16B LDS chunks -> one 16-element bf16 fragment (2x ds_load_b128)
__device__ __forceinline__ v16bf load_frag(const unsigned short* p0,
                                           const unsigned short* p1) {
  v8bf lo = *(const v8bf*)p0;
  v8bf hi = *(const v8bf*)p1;
  return __builtin_shufflevector(lo, hi, 0, 1, 2, 3, 4, 5, 6, 7,
                                         8, 9, 10, 11, 12, 13, 14, 15);
}

// ---------------------------------------------------------------------------
// Generic batched GEMM:  C = alpha * A @ B (+bias) (+relu), bf16 WMMA, f32 acc
// A: MxK row-major (lda). B: logical KxN; transB=0 -> row-major KxN (ldb),
// transB=1 -> stored NxK row-major (ldb). Batch offset for z (grid.z):
//   off = (z/divz)*Outer + (z%divz)*Inner   (per A/B/C)
// Tile: 64x64x64, 128 threads (4 waves); wave w owns rows [16w,16w+16);
// 8 WMMAs (2 K-subtiles x 4 N-subtiles) per staged tile.
// Requirements (all true here): M%64==0, N%64==0, K%64==0, lda/ldb%4==0,
// base pointers 16B aligned.
// ---------------------------------------------------------------------------
__global__ __launch_bounds__(128) void gemm_bf16_kernel(
    const float* __restrict__ A, long long aOuter, long long aInner, int lda,
    const float* __restrict__ Bm, long long bOuter, long long bInner, int ldb,
    int transB,
    float* __restrict__ Cm, long long cOuter, long long cInner, int ldc,
    const float* __restrict__ bias,
    int M, int N, int K, float alpha, int relu, int divz)
{
  // row stride 72 ushorts = 144B: multiple of 16B (aligned b128 chunks),
  // not a multiple of 256B (spreads banks).
  __shared__ __align__(16) unsigned short As[64][72];   // [m][k]
  __shared__ __align__(16) unsigned short Bs[64][72];   // [n][k] (K-contiguous)

  const int tid  = threadIdx.x;
  const int z    = blockIdx.z;
  const float* Ap = A  + (long long)(z / divz) * aOuter + (long long)(z % divz) * aInner;
  const float* Bp = Bm + (long long)(z / divz) * bOuter + (long long)(z % divz) * bInner;
  float*       Cp = Cm + (long long)(z / divz) * cOuter + (long long)(z % divz) * cInner;

  const int row0 = blockIdx.y * 64;
  const int col0 = blockIdx.x * 64;
  const int lane = tid & 31;
  const int wave = tid >> 5;
  const int half = lane >> 4;
  const int l16  = lane & 15;

  v8f acc[4];
  acc[0] = v8f{}; acc[1] = v8f{}; acc[2] = v8f{}; acc[3] = v8f{};

  for (int k0 = 0; k0 < K; k0 += 64) {
    // ---- stage A tile (64 rows x 64 k): 1024 float4 tasks, 8 per thread ----
    #pragma unroll
    for (int p = 0; p < 8; ++p) {
      int pos = tid + p * 128;          // 0..1023
      int m = pos >> 4;                 // 16 float4 per row
      int q = pos & 15;                 // kk = q*4
      float4 f = *(const float4*)&Ap[(size_t)(row0 + m) * lda + k0 + q * 4];
      store4bf(&As[m][q * 4], f);
    }
    // ---- stage B tile into [n][k] ----
    if (transB) {
      // source rows are K-contiguous: float4 loads straight in
      #pragma unroll
      for (int p = 0; p < 8; ++p) {
        int pos = tid + p * 128;
        int n = pos >> 4;
        int q = pos & 15;
        float4 f = *(const float4*)&Bp[(size_t)(col0 + n) * ldb + k0 + q * 4];
        store4bf(&Bs[n][q * 4], f);
      }
    } else {
      // source rows are N-contiguous: 4 lane-coalesced b32 loads per task,
      // packed into one b64 LDS store
      #pragma unroll
      for (int p = 0; p < 8; ++p) {
        int pos = tid + p * 128;
        int n  = pos & 63;
        int kg = pos >> 6;              // kk = kg*4
        const float* src = &Bp[(size_t)(k0 + kg * 4) * ldb + col0 + n];
        float4 f;
        f.x = src[0];
        f.y = src[(size_t)ldb];
        f.z = src[(size_t)ldb * 2];
        f.w = src[(size_t)ldb * 3];
        store4bf(&Bs[n][kg * 4], f);
      }
    }
    // prefetch next K tile of A (global_prefetch_b8)
    if (k0 + 64 < K)
      __builtin_prefetch(&Ap[(size_t)(row0 + (tid >> 1)) * lda + k0 + 64], 0, 1);
    __syncthreads();

    // ---- 2 K-subtiles x (1 A frag + 4 B frags -> 4 WMMA) ----
    #pragma unroll
    for (int ks = 0; ks < 2; ++ks) {
      const unsigned short* arow = &As[(wave << 4) + l16][ks * 32];
      // A fragment: elements j<8 <-> kk = half*8+j ; j>=8 <-> kk = 16+half*8+(j-8)
      v16bf afrag = load_frag(arow + (half << 3), arow + 16 + (half << 3));
      #pragma unroll
      for (int nb = 0; nb < 4; ++nb) {
        const unsigned short* brow = &Bs[nb * 16 + l16][ks * 32];
        // B fragment: elements j<8 <-> kk = half*8+j ; j>=8 <-> kk = 16+half*8+(j-8)
        v16bf bfrag = load_frag(brow + (half << 3), brow + 16 + (half << 3));
        acc[nb] = __builtin_amdgcn_wmma_f32_16x16x32_bf16(
            false, afrag, false, bfrag, (short)0, acc[nb], false, false);
      }
    }
    __syncthreads();
  }

  // ---- epilogue: alpha, bias, relu ----
  #pragma unroll
  for (int nb = 0; nb < 4; ++nb) {
    int gn = col0 + nb * 16 + l16;
    float bv = bias ? bias[gn] : 0.0f;
    #pragma unroll
    for (int rr = 0; rr < 8; ++rr) {
      int gm = row0 + (wave << 4) + (half << 3) + rr;
      float val = alpha * acc[nb][rr] + bv;
      if (relu) val = fmaxf(val, 0.0f);
      Cp[(size_t)gm * ldc + gn] = val;
    }
  }
}

// ---------------------------------------------------------------------------
// LayerNorm with residual: out = LN(xin + add) * g + b, row width = CC(=512)
// one block (256 thr) per row; safe in-place (reads precede writes).
// ---------------------------------------------------------------------------
__global__ __launch_bounds__(256) void ln_kernel(
    const float* __restrict__ xin, const float* __restrict__ add,
    const float* __restrict__ g, const float* __restrict__ bb,
    float* __restrict__ out)
{
  __shared__ float sm[256];
  const size_t row = blockIdx.x;
  const int tid = threadIdx.x;
  const float* xp = xin + row * CC;
  const float* ap = add + row * CC;
  float v0 = xp[tid] + ap[tid];
  float v1 = xp[tid + 256] + ap[tid + 256];

  sm[tid] = v0 + v1; __syncthreads();
  for (int s = 128; s > 0; s >>= 1) { if (tid < s) sm[tid] += sm[tid + s]; __syncthreads(); }
  float mu = sm[0] * (1.0f / CC); __syncthreads();

  float d0 = v0 - mu, d1 = v1 - mu;
  sm[tid] = d0 * d0 + d1 * d1; __syncthreads();
  for (int s = 128; s > 0; s >>= 1) { if (tid < s) sm[tid] += sm[tid + s]; __syncthreads(); }
  float rs = rsqrtf(sm[0] * (1.0f / CC) + 1e-5f);

  out[row * CC + tid]       = d0 * rs * g[tid]       + bb[tid];
  out[row * CC + tid + 256] = d1 * rs * g[tid + 256] + bb[tid + 256];
}

// ---------------------------------------------------------------------------
// In-place softmax over attention score rows; optional segment mask computed
// on the fly (seg!=null): add -1e9 where seg[b,t] != seg[b,s].
// grid.x = B*H*T rows of width TT; 256 threads x 4 elems.
// ---------------------------------------------------------------------------
__global__ __launch_bounds__(256) void attn_softmax_kernel(
    float* __restrict__ p, const int* __restrict__ seg)
{
  __shared__ float sm[256];
  const int row = blockIdx.x;
  const int tid = threadIdx.x;
  const int b = row / (HH * TT);
  const int t = row % TT;
  float* sp = p + (size_t)row * TT;
  const int st = seg ? seg[b * TT + t] : 0;

  float v[4]; float mx = -3.0e38f;
  #pragma unroll
  for (int e = 0; e < 4; ++e) {
    int col = tid + e * 256;
    float val = sp[col];
    if (seg && seg[b * TT + col] != st) val += -1e9f;
    v[e] = val; mx = fmaxf(mx, val);
  }
  sm[tid] = mx; __syncthreads();
  for (int s = 128; s > 0; s >>= 1) { if (tid < s) sm[tid] = fmaxf(sm[tid], sm[tid + s]); __syncthreads(); }
  mx = sm[0]; __syncthreads();

  float lsum = 0.0f;
  #pragma unroll
  for (int e = 0; e < 4; ++e) { v[e] = __expf(v[e] - mx); lsum += v[e]; }
  sm[tid] = lsum; __syncthreads();
  for (int s = 128; s > 0; s >>= 1) { if (tid < s) sm[tid] += sm[tid + s]; __syncthreads(); }
  float inv = 1.0f / sm[0];
  #pragma unroll
  for (int e = 0; e < 4; ++e) sp[tid + e * 256] = v[e] * inv;
}

// r[b,s] = (1/H) * sum_{h,t} p[b,h,t,s]   grid (BB, TT/256), 256 thr
__global__ __launch_bounds__(256) void probsum_kernel(
    const float* __restrict__ p, float* __restrict__ r)
{
  const int b = blockIdx.x;
  const int s = blockIdx.y * 256 + threadIdx.x;
  const float* bp = p + (size_t)b * HH * TT * TT;
  float acc = 0.0f;
  for (int ht = 0; ht < HH * TT; ++ht) acc += bp[(size_t)ht * TT + s];
  r[b * TT + s] = acc * (1.0f / HH);
}

// per-batch min/max normalize to [0,1]; one block per b
__global__ __launch_bounds__(256) void normalize_kernel(
    const float* __restrict__ r, float* __restrict__ w)
{
  __shared__ float smn[256]; __shared__ float smx[256];
  const int b = blockIdx.x, tid = threadIdx.x;
  const float* rp = r + b * TT;
  float mn = 3.0e38f, mx = -3.0e38f;
  #pragma unroll
  for (int e = 0; e < 4; ++e) {
    float v = rp[tid + e * 256]; mn = fminf(mn, v); mx = fmaxf(mx, v);
  }
  smn[tid] = mn; smx[tid] = mx; __syncthreads();
  for (int s = 128; s > 0; s >>= 1) {
    if (tid < s) { smn[tid] = fminf(smn[tid], smn[tid + s]); smx[tid] = fmaxf(smx[tid], smx[tid + s]); }
    __syncthreads();
  }
  float lo = smn[0]; float inv = 1.0f / fmaxf(smx[0] - lo, 1e-12f);
  #pragma unroll
  for (int e = 0; e < 4; ++e)
    w[b * TT + tid + e * 256] = (rp[tid + e * 256] - lo) * inv;
}

// row softmax (width TT) in->out; one block per row (grid = BB), for aw
__global__ __launch_bounds__(256) void vec_softmax_kernel(
    const float* __restrict__ in, float* __restrict__ out)
{
  __shared__ float sm[256];
  const int b = blockIdx.x, tid = threadIdx.x;
  const float* ip = in + b * TT;
  float v[4]; float mx = -3.0e38f;
  #pragma unroll
  for (int e = 0; e < 4; ++e) { v[e] = ip[tid + e * 256]; mx = fmaxf(mx, v[e]); }
  sm[tid] = mx; __syncthreads();
  for (int s = 128; s > 0; s >>= 1) { if (tid < s) sm[tid] = fmaxf(sm[tid], sm[tid + s]); __syncthreads(); }
  mx = sm[0]; __syncthreads();
  float lsum = 0.0f;
  #pragma unroll
  for (int e = 0; e < 4; ++e) { v[e] = __expf(v[e] - mx); lsum += v[e]; }
  sm[tid] = lsum; __syncthreads();
  for (int s = 128; s > 0; s >>= 1) { if (tid < s) sm[tid] += sm[tid + s]; __syncthreads(); }
  float inv = 1.0f / sm[0];
  #pragma unroll
  for (int e = 0; e < 4; ++e) out[b * TT + tid + e * 256] = v[e] * inv;
}

// sequential segment scan (faithful to reference lax.scan); 1 lane per batch
__global__ void seg_kernel(const float* __restrict__ w, int* __restrict__ seg)
{
  const int b = threadIdx.x;
  if (b >= BB) return;
  const float* wp = w + b * TT;
  int start = 0, id = 0;
  seg[b * TT] = 0;
  bool prev = wp[0] >= 0.5f;
  for (int t = 1; t < TT; ++t) {
    bool cur = wp[t] >= 0.5f;
    bool close = (cur != prev) && (start + 1 != t);
    if (close) { start = t; ++id; }
    seg[b * TT + t] = id;
    prev = cur;
  }
}

// wm[b,i,t] = (seg[b,t]==i)   flat over B*T*T
__global__ __launch_bounds__(256) void wm_kernel(
    const int* __restrict__ seg, float* __restrict__ wm)
{
  const int idx = blockIdx.x * 256 + threadIdx.x;
  const int t = idx & (TT - 1);
  const int i = (idx >> 10) & (TT - 1);
  const int b = idx >> 20;
  wm[idx] = (seg[b * TT + t] == i) ? 1.0f : 0.0f;
}

// zero word-token rows out[b, 0..T, :]
__global__ __launch_bounds__(256) void zero_wt_kernel(float* __restrict__ out)
{
  const int idx = blockIdx.x * 256 + threadIdx.x;
  const int c = idx & (CC - 1);
  const int t = (idx >> 9) & (TT - 1);
  const int b = idx >> 19;
  out[((size_t)b * (2 * TT) + t) * CC + c] = 0.0f;
}

// word_tokens[b, seg[b,t], c] += out_d[b,t,c] * aw[b,t]
__global__ __launch_bounds__(256) void scatter_wt_kernel(
    const float* __restrict__ xd, const float* __restrict__ aw,
    const int* __restrict__ seg, float* __restrict__ out)
{
  const int idx = blockIdx.x * 256 + threadIdx.x;
  const int c = idx & (CC - 1);
  const int t = (idx >> 9) & (TT - 1);
  const int b = idx >> 19;
  const float val = xd[idx] * aw[b * TT + t];
  const int w = seg[b * TT + t];
  atomicAdd(&out[((size_t)b * (2 * TT) + w) * CC + c], val);
}

// out[b, T+t, c] = x[b,t,c]
__global__ __launch_bounds__(256) void copy_x_kernel(
    const float* __restrict__ x, float* __restrict__ out)
{
  const int idx = blockIdx.x * 256 + threadIdx.x;
  const int c = idx & (CC - 1);
  const int t = (idx >> 9) & (TT - 1);
  const int b = idx >> 19;
  out[((size_t)b * (2 * TT) + TT + t) * CC + c] = x[idx];
}

// ---------------------------------------------------------------------------
extern "C" void kernel_launch(void* const* d_in, const int* in_sizes, int n_in,
                              void* d_out, int out_size, void* d_ws, size_t ws_size,
                              hipStream_t stream) {
  (void)in_sizes; (void)n_in; (void)out_size; (void)ws_size;

  const float* x    = (const float*)d_in[0];
  const float* Wqkv = (const float*)d_in[1];
  const float* bqkv = (const float*)d_in[2];
  const float* Wo   = (const float*)d_in[3];
  const float* bo   = (const float*)d_in[4];
  const float* W1   = (const float*)d_in[5];
  const float* b1   = (const float*)d_in[6];
  const float* W2   = (const float*)d_in[7];
  const float* b2   = (const float*)d_in[8];
  const float* ln_g = (const float*)d_in[9];
  const float* ln_b = (const float*)d_in[10];
  float* out = (float*)d_out;

  float* ws = (float*)d_ws;
  size_t o = 0;
  auto alloc = [&](size_t n) { float* p = ws + o; o += n; return p; };
  const size_t BTC = (size_t)BB * TT * CC;
  float* qb  = alloc(BTC);
  float* kb  = alloc(BTC);
  float* vb  = alloc(BTC);
  float* sc  = alloc((size_t)BB * HH * TT * TT);
  float* ob  = alloc(BTC);
  float* ab  = alloc(BTC);
  float* ffb = alloc((size_t)BB * TT * FF);
  float* xv  = alloc(BTC);
  float* xd  = alloc(BTC);
  float* rb  = alloc((size_t)BB * TT);
  float* wn  = alloc((size_t)BB * TT);
  float* awb = alloc((size_t)BB * TT);
  int*   segb = (int*)alloc((size_t)BB * TT);

  auto gemm = [&](const float* A, long long aO, long long aI, int lda,
                  const float* Bm, long long bO, long long bI, int ldb, int tB,
                  float* Cm, long long cO, long long cI, int ldc,
                  const float* bias, int M, int N, int K, float alpha, int relu,
                  int divz, int batch) {
    dim3 g(N / 64, M / 64, batch);
    gemm_bf16_kernel<<<g, dim3(128, 1, 1), 0, stream>>>(
        A, aO, aI, lda, Bm, bO, bI, ldb, tB, Cm, cO, cI, ldc,
        bias, M, N, K, alpha, relu, divz);
  };

  const int MBT = BB * TT; // 4096 rows for dense GEMMs

  auto run_tf = [&](int s, const float* xin, float* xcur, const int* segp, float* rout) {
    const float* cur = xin;
    for (int l = 0; l < 2; ++l) {
      const int pl = s * 2 + l;
      // QKV projections (3x dense C->C)
      float* dsts[3] = {qb, kb, vb};
      for (int k3 = 0; k3 < 3; ++k3) {
        gemm(cur, 0, 0, CC,
             Wqkv + ((size_t)pl * 3 + k3) * CC * CC, 0, 0, CC, 0,
             dsts[k3], 0, 0, CC,
             bqkv + ((size_t)pl * 3 + k3) * CC,
             MBT, CC, CC, 1.0f, 0, 1, 1);
      }
      // scores = q @ k^T / sqrt(dh)   (batched over B*H)
      gemm(qb, (long long)TT * CC, DH, CC,
           kb, (long long)TT * CC, DH, CC, 1,
           sc, (long long)HH * TT * TT, (long long)TT * TT, TT,
           nullptr, TT, TT, DH, 0.125f, 0, HH, BB * HH);
      // softmax (+ on-the-fly segment mask for word stream)
      attn_softmax_kernel<<<dim3(BB * HH * TT), dim3(256), 0, stream>>>(sc, segp);
      if (l == 0 && rout)
        probsum_kernel<<<dim3(BB, TT / 256), dim3(256), 0, stream>>>(sc, rout);
      // o = p @ v  (batched), written back into head-interleaved (B,T,C)
      gemm(sc, (long long)HH * TT * TT, (long long)TT * TT, TT,
           vb, (long long)TT * CC, DH, CC, 0,
           ob, (long long)TT * CC, DH, CC,
           nullptr, TT, DH, TT, 1.0f, 0, HH, BB * HH);
      // out projection
      gemm(ob, 0, 0, CC, Wo + (size_t)pl * CC * CC, 0, 0, CC, 0,
           ab, 0, 0, CC, bo + (size_t)pl * CC, MBT, CC, CC, 1.0f, 0, 1, 1);
      // x = LN(x + attn)
      ln_kernel<<<dim3(MBT), dim3(256), 0, stream>>>(
          cur, ab, ln_g + ((size_t)pl * 2) * CC, ln_b + ((size_t)pl * 2) * CC, xcur);
      // FFN: relu(x@W1+b1)@W2+b2
      gemm(xcur, 0, 0, CC, W1 + (size_t)pl * CC * FF, 0, 0, FF, 0,
           ffb, 0, 0, FF, b1 + (size_t)pl * FF, MBT, FF, CC, 1.0f, 1, 1, 1);
      gemm(ffb, 0, 0, FF, W2 + (size_t)pl * FF * CC, 0, 0, CC, 0,
           ab, 0, 0, CC, b2 + (size_t)pl * CC, MBT, CC, FF, 1.0f, 0, 1, 1);
      // x = LN(x + ffn)
      ln_kernel<<<dim3(MBT), dim3(256), 0, stream>>>(
          xcur, ab, ln_g + ((size_t)pl * 2 + 1) * CC, ln_b + ((size_t)pl * 2 + 1) * CC, xcur);
      cur = xcur;
    }
  };

  // verbal transformer on x -> xv, capture layer-0 head-mean prob column sums
  run_tf(0, x, xv, nullptr, rb);
  // w = normalize(r); segment ids from thresholded w
  normalize_kernel<<<dim3(BB), dim3(256), 0, stream>>>(rb, wn);
  seg_kernel<<<dim3(1), dim3(32), 0, stream>>>(wn, segb);
  // word transformer on xv with segment mask -> xd, capture its layer-0 sums
  run_tf(1, xv, xd, segb, rb);
  // aw = softmax(r)
  vec_softmax_kernel<<<dim3(BB), dim3(256), 0, stream>>>(rb, awb);

  // outputs: [concat(word_tokens, x) | wm]
  float* wm_out = out + (size_t)BB * 2 * TT * CC;
  wm_kernel<<<dim3((BB * TT * TT) / 256), dim3(256), 0, stream>>>(segb, wm_out);
  zero_wt_kernel<<<dim3((int)(BTC / 256)), dim3(256), 0, stream>>>(out);
  scatter_wt_kernel<<<dim3((int)(BTC / 256)), dim3(256), 0, stream>>>(xd, awb, segb, out);
  copy_x_kernel<<<dim3((int)(BTC / 256)), dim3(256), 0, stream>>>(x, out);
}
